// Head_781684048730
// MI455X (gfx1250) — matrix operand order repaired and checked by reading, not compiled, not run on hardware
//
#include <hip/hip_runtime.h>
#include <hip/hip_bf16.h>
#include <math.h>

// Problem constants (from reference): B=256, T=256, C=384, H=64
#define BB 256
#define TT 256
#define CC 384
#define HH 64

typedef __attribute__((ext_vector_type(16))) _Float16 v16h;
typedef __attribute__((ext_vector_type(8)))  _Float16 v8h;
typedef __attribute__((ext_vector_type(8)))  float    v8f;
typedef __attribute__((ext_vector_type(4)))  float    f4;

static __device__ __forceinline__ v16h cat8(v8h lo, v8h hi) {
    return __builtin_shufflevector(lo, hi, 0,1,2,3,4,5,6,7,8,9,10,11,12,13,14,15);
}

static __device__ __forceinline__ v8f wmma_f16(v16h a, v16h b, v8f c) {
    return __builtin_amdgcn_wmma_f32_16x16x32_f16(false, a, false, b, (short)0, c,
                                                  false, false);
}

// ---------------------------------------------------------------------------
// Kernel 1: fused QKV projection, single pass over x.
// x: [B*T, 384] f32.  Outputs: q,k as [B*T, 64] f16; v stored transposed as
// vT [B][64][256] f16 so the attention kernel's PV B-fragments are contiguous.
// All three weights pre-swizzled into WMMA-B fragment layout in LDS (144 KB).
// LDS fragments are ordered kc-major so the 12 fragments consumed per K-chunk
// iteration are one contiguous 12 KB window -> all DS immediate offsets fit
// the 16-bit field (no per-fragment address VALU, no WMMA WAR hazard NOPs).
// ---------------------------------------------------------------------------
__global__ __launch_bounds__(256) void qkv_proj_kernel(
    const float* __restrict__ x,
    const float* __restrict__ Wq, const float* __restrict__ Wk,
    const float* __restrict__ Wv,
    _Float16* __restrict__ qo, _Float16* __restrict__ ko,
    _Float16* __restrict__ vto)
{
    // 12 k-chunks * 3 weights * 4 n-tiles * 32 lanes * 16 halfs = 73728 f16
    __shared__ _Float16 ldsW[12 * 3 * 4 * 32 * 16];   // 144 KB

    const int tid  = threadIdx.x;
    const int lane = tid & 31;
    const int wv   = tid >> 5;
    const int h    = lane >> 4;   // lane half (wave32 fragment layout)
    const int l16  = lane & 15;
    const int rowBase = blockIdx.x * 128 + wv * 16;

    const float* Ws[3] = { Wq, Wk, Wv };

    // Stage Wq|Wk|Wv into LDS in WMMA-B fragment order, kc-major:
    // frag index = (kc*3 + w)*4 + nt;  element j of lane L:
    //   k = kc*32 + j + 8*(j>=8) + 8*(L>>4),  n = nt*16 + (L&15)
    for (int e = tid; e < 12 * 3 * 4 * 512; e += 256) {
        int frag = e >> 9;            // 512 f16 per fragment
        int rem  = e & 511;
        int ln   = rem >> 4;
        int j    = rem & 15;
        int nt   = frag & 3;
        int t    = frag >> 2;         // kc*3 + w
        int kc   = t / 3;
        int w    = t - kc * 3;
        int kk   = kc * 32 + j + 8 * (j >= 8) + 8 * (ln >> 4);
        int nn   = nt * 16 + (ln & 15);
        ldsW[e]  = (_Float16)Ws[w][kk * HH + nn];
    }
    __syncthreads();

    v8f c[3][4] = {};
    const float* xrow = x + (size_t)(rowBase + l16) * CC;

    for (int kc = 0; kc < 12; ++kc) {
        // A fragment: element j -> dim = kc*32 + j + 8*(j>=8) + 8*h
        const float* p0 = xrow + kc * 32 + 8 * h;
        f4 a0 = *(const f4*)(p0);
        f4 a1 = *(const f4*)(p0 + 4);
        f4 a2 = *(const f4*)(p0 + 16);
        f4 a3 = *(const f4*)(p0 + 20);
        v16h a;
        #pragma unroll
        for (int u = 0; u < 4; ++u) {
            a[u]      = (_Float16)a0[u];
            a[4 + u]  = (_Float16)a1[u];
            a[8 + u]  = (_Float16)a2[u];
            a[12 + u] = (_Float16)a3[u];
        }
        #pragma unroll
        for (int w = 0; w < 3; ++w) {
            #pragma unroll
            for (int nt = 0; nt < 4; ++nt) {
                // kc-major: all 12 fragments of this kc within 12 KB window
                const int fb = (((kc * 3 + w) * 4) + nt) * 512 + lane * 16;
                v8h blo = *(const v8h*)&ldsW[fb];
                v8h bhi = *(const v8h*)&ldsW[fb + 8];
                c[w][nt] = wmma_f16(a, cat8(blo, bhi), c[w][nt]);
            }
        }
    }

    // ---- branch-free epilogue --------------------------------------------
    // D layout: vgpr i <-> row i + 8h, lane&15 <-> col within n-tile.
    #pragma unroll
    for (int nt = 0; nt < 4; ++nt) {
        const int col = nt * 16 + l16;
        #pragma unroll
        for (int i = 0; i < 8; ++i) {
            const int rr = rowBase + i + 8 * h;
            qo[(size_t)rr * HH + col] = (_Float16)c[0][nt][i];
            ko[(size_t)rr * HH + col] = (_Float16)c[1][nt][i];
        }
        // V transposed: for fixed (lane, nt) the 8 rows are consecutive t's
        // -> one packed 16-byte store.
        v8h pk;
        #pragma unroll
        for (int i = 0; i < 8; ++i) pk[i] = (_Float16)c[2][nt][i];
        const int bb = rowBase >> 8;                 // batch (tile never spans)
        const int t0 = (rowBase & 255) + 8 * h;      // first seq pos, 8-aligned
        *(v8h*)&vto[((size_t)bb * HH + col) * TT + t0] = pk;
    }
}

// ---------------------------------------------------------------------------
// Kernel 2: flash-style causal attention per batch.
// grid (B, 2); 8 waves/block; wave handles 16 query rows.
// Per 32-key block: 4 WMMA for S = Q K^T, online softmax, LDS layout-swap of
// the P tile (C/D -> A fragment), 4 WMMA for O += P V.
// ---------------------------------------------------------------------------
__global__ __launch_bounds__(256) void attn_kernel(
    const _Float16* __restrict__ q, const _Float16* __restrict__ k,
    const _Float16* __restrict__ vt, float* __restrict__ out)
{
    __shared__ _Float16 ldsP[8 * 512];   // 1 KB per wave P tile (16x32 f16)

    const int tid  = threadIdx.x;
    const int lane = tid & 31;
    const int wv   = tid >> 5;
    const int h    = lane >> 4;
    const int l16  = lane & 15;
    const int b    = blockIdx.x;
    const int qBase = blockIdx.y * 128 + wv * 16;
    const float scale = 0.05103103630798288f;   // 384^-0.5 (C, per reference)

    _Float16* P = ldsP + wv * 512;

    // Q A-fragments (16 rows x 64 dims -> two K=32 chunks)
    const _Float16* qrow = q + (size_t)(b * TT + qBase + l16) * HH;
    v16h aQ[2];
    #pragma unroll
    for (int kc = 0; kc < 2; ++kc) {
        v8h lo = *(const v8h*)(qrow + kc * 32 + 16 * h);
        v8h hi = *(const v8h*)(qrow + kc * 32 + 16 * h + 8);
        aQ[kc] = cat8(lo, hi);
    }

    v8f o[4] = { v8f{}, v8f{}, v8f{}, v8f{} };
    float mrow[8], lrow[8];
    #pragma unroll
    for (int i = 0; i < 8; ++i) { mrow[i] = -INFINITY; lrow[i] = 0.0f; }

    for (int j0 = 0; j0 <= qBase + 15; j0 += 32) {
        // ---- S = Q K^T for two 16-key tiles -------------------------------
        v8f s[2];
        #pragma unroll
        for (int kt = 0; kt < 2; ++kt) {
            // B element j: dim = kc*32 + j + 16h, key = j0 + kt*16 + (L&15)
            const _Float16* krow =
                k + (size_t)(b * TT + j0 + kt * 16 + l16) * HH;
            v8f acc{};
            #pragma unroll
            for (int kc = 0; kc < 2; ++kc) {
                v8h lo = *(const v8h*)(krow + kc * 32 + 16 * h);
                v8h hi = *(const v8h*)(krow + kc * 32 + 16 * h + 8);
                acc = wmma_f16(aQ[kc], cat8(lo, hi), acc);
            }
            s[kt] = acc;
        }

        // ---- causal mask + online softmax ---------------------------------
        #pragma unroll
        for (int i = 0; i < 8; ++i) {
            int row = qBase + i + 8 * h;
            int k0  = j0 + l16;
            int k1  = j0 + 16 + l16;
            float s0 = (k0 <= row) ? s[0][i] * scale : -INFINITY;
            float s1 = (k1 <= row) ? s[1][i] * scale : -INFINITY;

            float m = fmaxf(s0, s1);
            m = fmaxf(m, __shfl_xor(m, 1));
            m = fmaxf(m, __shfl_xor(m, 2));
            m = fmaxf(m, __shfl_xor(m, 4));
            m = fmaxf(m, __shfl_xor(m, 8));

            float mn   = fmaxf(mrow[i], m);
            float corr = __expf(mrow[i] - mn);
            mrow[i] = mn;

            float p0 = __expf(s0 - mn);
            float p1 = __expf(s1 - mn);
            float rs = p0 + p1;
            rs += __shfl_xor(rs, 1);
            rs += __shfl_xor(rs, 2);
            rs += __shfl_xor(rs, 4);
            rs += __shfl_xor(rs, 8);
            lrow[i] = lrow[i] * corr + rs;

            #pragma unroll
            for (int nt = 0; nt < 4; ++nt) o[nt][i] *= corr;

            // P tile row-major [16 rows][32 keys] in per-wave LDS
            P[(i + 8 * h) * 32 + l16]      = (_Float16)p0;
            P[(i + 8 * h) * 32 + 16 + l16] = (_Float16)p1;
        }

        // ---- layout swap: P tile C/D layout -> A fragment -----------------
        // A element j: key = j + 16h, row = L&15
        v8h plo = *(const v8h*)&P[l16 * 32 + 16 * h];
        v8h phi = *(const v8h*)&P[l16 * 32 + 16 * h + 8];
        v16h aP = cat8(plo, phi);

        // ---- O += P V  (V pre-transposed: vt[b][dim][key]) ----------------
        #pragma unroll
        for (int nt = 0; nt < 4; ++nt) {
            // B element j: key = j0 + j + 16h, dim = nt*16 + (L&15)
            const _Float16* vrow =
                vt + ((size_t)b * HH + nt * 16 + l16) * TT + j0 + 16 * h;
            v8h vlo = *(const v8h*)(vrow);
            v8h vhi = *(const v8h*)(vrow + 8);
            o[nt] = wmma_f16(aP, cat8(vlo, vhi), o[nt]);
        }
    }

    // ---- epilogue: normalize, write f32 -----------------------------------
    #pragma unroll
    for (int i = 0; i < 8; ++i) {
        float inv = 1.0f / lrow[i];
        int t = qBase + i + 8 * h;
        float* orow = out + (size_t)(b * TT + t) * HH;
        #pragma unroll
        for (int nt = 0; nt < 4; ++nt)
            orow[nt * 16 + l16] = o[nt][i] * inv;
    }
}

// ---------------------------------------------------------------------------
extern "C" void kernel_launch(void* const* d_in, const int* in_sizes, int n_in,
                              void* d_out, int out_size, void* d_ws,
                              size_t ws_size, hipStream_t stream) {
    (void)in_sizes; (void)n_in; (void)out_size; (void)ws_size;

    const float* x  = (const float*)d_in[0];
    const float* Wq = (const float*)d_in[1];
    const float* Wk = (const float*)d_in[2];
    const float* Wv = (const float*)d_in[3];
    float* out = (float*)d_out;

    // Workspace: q, k as [B*T,64] f16; vT as [B][64][256] f16 (8 MB each)
    const size_t qkv_elems = (size_t)BB * TT * HH;
    _Float16* qb  = (_Float16*)d_ws;
    _Float16* kb  = qb + qkv_elems;
    _Float16* vtb = kb + qkv_elems;

    // 512 workgroups x 128 rows = 65536 rows
    qkv_proj_kernel<<<dim3((BB * TT) / 128), dim3(256), 0, stream>>>(
        x, Wq, Wk, Wv, qb, kb, vtb);

    attn_kernel<<<dim3(BB, 2), dim3(256), 0, stream>>>(qb, kb, vtb, out);
}
